// SimpleLSTM_35038343200792
// MI455X (gfx1250) — matrix operand order repaired and checked by reading, not compiled, not run on hardware
//
#include <hip/hip_runtime.h>
#include <hip/hip_bf16.h>
#include <stdint.h>

// ---------------------------------------------------------------------------
// SimpleLSTM on MI455X (gfx1250, wave32, WMMA)
//   prep   : one-pass f32->bf16 conversion of emb and W_ih into workspace
//   phase 1: fused embedding-gather + x@W_ih^T (+bias) -> xg [T][B][4H] (f32)
//            W_ih register-resident per wave (loaded once), embedding rows
//            staged through LDS; bf16 WMMA, no conversions in hot loop
//   phase 2: persistent-weight LSTM scan, W_hh held in VGPRs as bf16 tiles,
//            8 blocks x 16 waves, 8 WMMA/wave/step, prefetch of next slab
//   phase 3: tiny FC head from f32 h in LDS
// ---------------------------------------------------------------------------

typedef __attribute__((ext_vector_type(16))) __bf16 v16bf;
typedef __attribute__((ext_vector_type(8)))  __bf16 bf16x8;
typedef __attribute__((ext_vector_type(8)))  float  v8f;
typedef __attribute__((ext_vector_type(4)))  float  f32x4;

#define VOCAB   50257
#define BATCH   128
#define SEQ     1024
#define EMBED   256
#define HIDDEN  128
#define GATES   512                 // 4*HIDDEN, torch order: i, f, g, o
#define ROWS    (BATCH * SEQ)       // 131072
#define MT_PER_BLK 16               // M tiles per block in kernel 1

__device__ __forceinline__ float sigmoidf_fast(float x) {
    return 1.0f / (1.0f + __expf(-x));
}

// ---------------------------------------------------------------------------
// Prep: f32 -> bf16 stream conversion (count must be a multiple of 4)
// ---------------------------------------------------------------------------
__global__ __launch_bounds__(256)
void cvt_f32_bf16(const float* __restrict__ src, __bf16* __restrict__ dst,
                  int count4)
{
    const int i = blockIdx.x * 256 + threadIdx.x;
    if (i < count4) {
        f32x4 v = *(const f32x4*)(src + (size_t)i * 4);
        __bf16 o[4];
        #pragma unroll
        for (int j = 0; j < 4; ++j) o[j] = (__bf16)v[j];
        *(uint64_t*)(dst + (size_t)i * 4) = *(const uint64_t*)o;
    }
}

// ---------------------------------------------------------------------------
// Kernel 1: xg[t, b, g] = sum_e emb[x[b,t], e] * W_ih[g, e] + (b_ih[g]+b_hh[g])
// grid: ROWS/256 = 512 blocks, 1024 threads (32 waves).
// Wave w owns the 16-column tile [w*16, w*16+16) and keeps its whole K=256
// slice of W_ih in VGPRs (8 x v16bf).  Block loops over 16 M tiles; the
// 16x256 bf16 embedding tile is staged in LDS once per M tile.
// ---------------------------------------------------------------------------
__global__ __launch_bounds__(1024)
void lstm_xg_wmma(const long long* __restrict__ x,
                  const __bf16*    __restrict__ emb_bf,
                  const __bf16*    __restrict__ wih_bf,
                  const float*     __restrict__ b_ih,
                  const float*     __restrict__ b_hh,
                  float*           __restrict__ xg)
{
    __shared__ __bf16 a_sh[16][EMBED];    // 8 KB staged A tile (16 rows x K=256)

    const int tid   = threadIdx.x;
    const int wave  = tid >> 5;           // 0..31 == n-tile index
    const int lane  = tid & 31;
    const int khalf = lane >> 4;          // 0: k 0-7,16-23 of each 32; 1: k 8-15,24-31
    const int lan16 = lane & 15;
    const int col   = wave * 16 + lan16;  // this lane's output column

    // ---- persistent B tiles: B[k][g] = W_ih[col, k], two 8-K runs per lane (k0, k0+16)
    v16bf Bt[8];
    #pragma unroll
    for (int ks = 0; ks < 8; ++ks) {
        const int kb = ks * 32 + khalf * 8;
        bf16x8 blo = *(const bf16x8*)(wih_bf + (size_t)col * EMBED + kb);
        bf16x8 bhi = *(const bf16x8*)(wih_bf + (size_t)col * EMBED + kb + 16);
        #pragma unroll
        for (int j = 0; j < 8; ++j) { Bt[ks][j] = blo[j]; Bt[ks][j + 8] = bhi[j]; }
    }
    const float bs = b_ih[col] + b_hh[col];

    const int rowblk = blockIdx.x * (MT_PER_BLK * 16);  // 256 consecutive rows
    const int srow   = tid >> 6;          // staging: 64 threads per row
    const int scol   = (tid & 63) * 4;    // 4 bf16 (8 B) per thread

    #pragma unroll 1
    for (int m = 0; m < MT_PER_BLK; ++m) {
        const int mbase = rowblk + m * 16;

        // ---- cooperative embedding gather into LDS (8 B per thread) ----
        {
            const long long tok = x[mbase + srow];
            *(uint64_t*)&a_sh[srow][scol] =
                *(const uint64_t*)(emb_bf + (size_t)tok * EMBED + scol);
        }
        __syncthreads();

        // ---- 8 WMMAs over K=256, A from LDS, B from registers ----
        v8f acc = {};
        #pragma unroll
        for (int ks = 0; ks < 8; ++ks) {
            const int k0 = ks * 32 + khalf * 8;
            const int k1 = ks * 32 + 16 + khalf * 8;
            bf16x8 lo = *(const bf16x8*)&a_sh[lan16][k0];
            bf16x8 hi = *(const bf16x8*)&a_sh[lan16][k1];
            v16bf a;
            #pragma unroll
            for (int j = 0; j < 8; ++j) { a[j] = lo[j]; a[j + 8] = hi[j]; }
            acc = __builtin_amdgcn_wmma_f32_16x16x32_bf16(
                      false, a, false, Bt[ks], (short)0, acc, false, false);
        }

        // ---- bias + store to [T][B][G]; D: VGPR r -> row = r + 8*khalf ----
        #pragma unroll
        for (int r = 0; r < 8; ++r) {
            const int row = mbase + r + khalf * 8;   // = b*SEQ + t
            const int b   = row >> 10;               // /SEQ
            const int t   = row & (SEQ - 1);
            xg[((size_t)t * BATCH + b) * GATES + col] = acc[r] + bs;
        }
        __syncthreads();                  // protect a_sh before next stage
    }
}

// ---------------------------------------------------------------------------
// Kernel 2: persistent LSTM scan. grid: BATCH/16 = 8 blocks, 512 threads
// (16 waves). Wave w owns gate cols [w*32, w*32+32); W_hh slice lives in
// VGPRs as bf16 for the whole kernel. c-state in registers (4 per thread).
// xg is [T][B][G]: each step reads one contiguous 32 KB tile per block,
// and prefetches the next step's tile.
// ---------------------------------------------------------------------------
__global__ __launch_bounds__(512)
void lstm_scan_wmma(const float* __restrict__ xg,
                    const float* __restrict__ W_hh,
                    const float* __restrict__ W_fc,
                    const float* __restrict__ b_fc,
                    float*       __restrict__ out)
{
    __shared__ __bf16 h_bf[16][HIDDEN];     //  4 KB: A-matrix for WMMA
    __shared__ float  h_f [16][HIDDEN];     //  8 KB: f32 h for final head
    __shared__ float  gate[16][GATES];      // 32 KB: step's gate pre-activations

    const int tid   = threadIdx.x;
    const int wave  = tid >> 5;             // 0..15
    const int lane  = tid & 31;
    const int khalf = lane >> 4;
    const int lan16 = lane & 15;
    const int bbase = blockIdx.x * 16;      // this block's 16 batch rows

    // ---- load persistent W_hh tiles: B[k][g] = W_hh[g, k], bf16 in VGPRs ----
    v16bf Wt[2][4];
    #pragma unroll
    for (int n = 0; n < 2; ++n) {
        const int col = wave * 32 + n * 16 + lan16;
        #pragma unroll
        for (int ks = 0; ks < 4; ++ks) {
            const int kb = ks * 32 + khalf * 8;
            const float* brow = W_hh + (size_t)col * HIDDEN + kb;
            f32x4 b0 = *(const f32x4*)(brow);
            f32x4 b1 = *(const f32x4*)(brow + 4);
            f32x4 b2 = *(const f32x4*)(brow + 16);
            f32x4 b3 = *(const f32x4*)(brow + 20);
            #pragma unroll
            for (int j = 0; j < 4; ++j) {
                Wt[n][ks][j]      = (__bf16)b0[j];
                Wt[n][ks][j + 4]  = (__bf16)b1[j];
                Wt[n][ks][j + 8]  = (__bf16)b2[j];
                Wt[n][ks][j + 12] = (__bf16)b3[j];
            }
        }
    }

    // ---- init h = c = 0 ----
    float c_reg[4] = {0.0f, 0.0f, 0.0f, 0.0f};
    for (int e = tid; e < 16 * HIDDEN; e += 512) {
        (&h_bf[0][0])[e] = (__bf16)0.0f;
        (&h_f [0][0])[e] = 0.0f;
    }
    __syncthreads();

    #pragma unroll 1
    for (int t = 0; t < SEQ; ++t) {
        const float* slab = xg + ((size_t)t * BATCH + bbase) * GATES;

        // prefetch next step's contiguous 32 KB tile (global_prefetch_b8)
        if (t + 1 < SEQ) {
            const char* nxt = (const char*)(slab + (size_t)BATCH * GATES);
            __builtin_prefetch(nxt + tid * 64, 0, 1);
        }

        // ---- A tiles (h, 16x32 bf16) from LDS: two contiguous 16B runs/lane ----
        v16bf A[4];
        #pragma unroll
        for (int ks = 0; ks < 4; ++ks) {
            const int k0 = ks * 32 + khalf * 8;
            const int k1 = ks * 32 + 16 + khalf * 8;
            bf16x8 lo = *(const bf16x8*)&h_bf[lan16][k0];
            bf16x8 hi = *(const bf16x8*)&h_bf[lan16][k1];
            #pragma unroll
            for (int j = 0; j < 8; ++j) { A[ks][j] = lo[j]; A[ks][j + 8] = hi[j]; }
        }

        // ---- gates tile = xg[t, :, cols] + h @ W_hh^T ----
        #pragma unroll
        for (int n = 0; n < 2; ++n) {
            const int col = wave * 32 + n * 16 + lan16;
            v8f acc;
            #pragma unroll
            for (int r = 0; r < 8; ++r) {
                const int b = r + khalf * 8;
                acc[r] = slab[(size_t)b * GATES + col];
            }
            #pragma unroll
            for (int ks = 0; ks < 4; ++ks)
                acc = __builtin_amdgcn_wmma_f32_16x16x32_bf16(
                          false, A[ks], false, Wt[n][ks], (short)0, acc, false, false);
            #pragma unroll
            for (int r = 0; r < 8; ++r)
                gate[r + khalf * 8][col] = acc[r];
        }
        __syncthreads();

        // ---- elementwise cell update (c in registers, 4 elems/thread) ----
        #pragma unroll
        for (int r = 0; r < 4; ++r) {
            const int e  = tid + 512 * r;   // 0..2047
            const int b  = e >> 7;          // batch row in tile
            const int hh = e & 127;         // hidden index
            const float ig = sigmoidf_fast(gate[b][hh]);
            const float fg = sigmoidf_fast(gate[b][HIDDEN + hh]);
            const float gg = tanhf(gate[b][2 * HIDDEN + hh]);
            const float og = sigmoidf_fast(gate[b][3 * HIDDEN + hh]);
            const float cn = fg * c_reg[r] + ig * gg;
            const float hn = og * tanhf(cn);
            c_reg[r] = cn;
            h_f [b][hh] = hn;
            h_bf[b][hh] = (__bf16)hn;
        }
        __syncthreads();
    }

    // ---- final head: out[b, cls] = h_T[b] . W_fc[cls] + b_fc[cls] ----
    if (tid < 32) {
        const int b   = tid & 15;
        const int cls = tid >> 4;
        float s = b_fc[cls];
        #pragma unroll 4
        for (int k = 0; k < HIDDEN; ++k)
            s += h_f[b][k] * W_fc[cls * HIDDEN + k];
        out[(bbase + b) * 2 + cls] = s;
    }
}

// ---------------------------------------------------------------------------
extern "C" void kernel_launch(void* const* d_in, const int* in_sizes, int n_in,
                              void* d_out, int out_size, void* d_ws, size_t ws_size,
                              hipStream_t stream) {
    (void)in_sizes; (void)n_in; (void)out_size; (void)ws_size;
    const long long* x    = (const long long*)d_in[0];   // int64 tokens [B, T]
    const float*     emb  = (const float*)d_in[1];       // [VOCAB, EMBED]
    const float*     W_ih = (const float*)d_in[2];       // [4H, E]
    const float*     W_hh = (const float*)d_in[3];       // [4H, H]
    const float*     b_ih = (const float*)d_in[4];       // [4H]
    const float*     b_hh = (const float*)d_in[5];       // [4H]
    const float*     W_fc = (const float*)d_in[6];       // [2, H]
    const float*     b_fc = (const float*)d_in[7];       // [2]
    float* out = (float*)d_out;                          // [B, 2]

    // workspace layout
    char*   ws     = (char*)d_ws;
    float*  xg     = (float*)ws;                               // 256 MiB [T][B][G]
    __bf16* emb_bf = (__bf16*)(ws + (size_t)ROWS * GATES * 4); // 24.5 MiB
    __bf16* wih_bf = emb_bf + (size_t)VOCAB * EMBED;           // 256 KiB

    // prep: one-pass bf16 conversions (bandwidth-trivial)
    {
        const int c4e = (VOCAB * EMBED) / 4;                   // divisible (EMBED=256)
        cvt_f32_bf16<<<(c4e + 255) / 256, 256, 0, stream>>>(emb, emb_bf, c4e);
        const int c4w = (GATES * EMBED) / 4;
        cvt_f32_bf16<<<(c4w + 255) / 256, 256, 0, stream>>>(W_ih, wih_bf, c4w);
    }

    lstm_xg_wmma<<<ROWS / (MT_PER_BLK * 16), 1024, 0, stream>>>(
        x, emb_bf, wih_bf, b_ih, b_hh, xg);
    lstm_scan_wmma<<<BATCH / 16, 512, 0, stream>>>(xg, W_hh, W_fc, b_fc, out);
}
